// NetVLAD_70231305224563
// MI455X (gfx1250) — matrix operand order repaired and checked by reading, not compiled, not run on hardware
//
#include <hip/hip_runtime.h>

#define NB 32
#define NL 4096
#define ND 1024
#define NK 64
#define LDSP 40   // padded LDS row stride (bf16 elements): multiple of 8 keeps b128 alignment
#define NEG_INF (-__builtin_inff())

typedef __attribute__((ext_vector_type(16))) __bf16 v16bf;
typedef __attribute__((ext_vector_type(2)))  __bf16 v2bf;
typedef __attribute__((ext_vector_type(8)))  float  v8f;

struct __attribute__((aligned(16))) U4 { unsigned int x, y, z, w; };
struct __attribute__((aligned(16))) F4 { float x, y, z, w; };

union Frag {
  v16bf v;
  U4 q[2];
};

// fp32 -> bf16 via hardware convert (RNE in HW), not integer emulation
__device__ __forceinline__ unsigned short f2bf(float f) {
  return __builtin_bit_cast(unsigned short, (__bf16)f);
}
__device__ __forceinline__ unsigned int pk2(float a, float b) {
#if __has_builtin(__builtin_amdgcn_cvt_pk_bf16_f32)
  v2bf p = __builtin_amdgcn_cvt_pk_bf16_f32(a, b);
  return __builtin_bit_cast(unsigned int, p);
#else
  v2bf p;
  p[0] = (__bf16)a;
  p[1] = (__bf16)b;
  return __builtin_bit_cast(unsigned int, p);
#endif
}
__device__ __forceinline__ v8f vzero() {
  v8f z;
  #pragma unroll
  for (int i = 0; i < 8; ++i) z[i] = 0.0f;
  return z;
}

// ---------------- Kernel 0: weights -> bf16, centroids -> transposed ----------------
__global__ __launch_bounds__(256) void k_prep(const float* __restrict__ aw,
                                              const float* __restrict__ cen,
                                              unsigned short* __restrict__ wbf,
                                              float* __restrict__ centT) {
  const int i = blockIdx.x * 256 + threadIdx.x;     // 0 .. K*D-1
  wbf[i] = f2bf(aw[i]);
  const int k = i >> 10;
  const int d = i & (ND - 1);
  centT[d * NK + k] = cen[i];
}

// ---------------- Kernel 1: logits[b][k][l] = feats · Wᵀ + bias, masked ----------------
__global__ __launch_bounds__(256) void k_logits(const float* __restrict__ feats,
                                                const int* __restrict__ lens,
                                                const unsigned short* __restrict__ wbf,
                                                const float* __restrict__ ab,
                                                float* __restrict__ logits) {
  const int b   = blockIdx.x >> 5;
  const int l0  = (blockIdx.x & 31) * 128;
  const int tid  = threadIdx.x;
  const int wave = tid >> 5;
  const int lane = tid & 31;

  __shared__ unsigned short lsA[2][128 * LDSP];     // double-buffered 128 L-rows x 32 d (bf16)

  const float* fb = feats + ((size_t)b * NL + l0) * ND;

  const int sr = tid >> 1;            // staging row 0..127
  const int sc = (tid & 1) * 16;      // staging col 0 or 16
  const float* src = fb + (size_t)sr * ND + sc;

  v8f acc[4];
  #pragma unroll
  for (int i = 0; i < 4; ++i) acc[i] = vzero();

  const int arow  = wave * 16 + (lane & 15);
  const int ahalf = (lane >> 4) * 8;   // A: K 0..7 / 8..15 (then +16)
  const int bhalf = (lane >> 4) * 8;   // B: K 0..7 / 8..15 (then +16)

  // pipeline prologue: tile 0 in registers
  F4 f0 = *(const F4*)(src + 0);
  F4 f1 = *(const F4*)(src + 4);
  F4 f2 = *(const F4*)(src + 8);
  F4 f3 = *(const F4*)(src + 12);

  for (int d0 = 0; d0 < ND; d0 += 32) {
    unsigned short* buf = lsA[(d0 >> 5) & 1];
    U4 p0 = { pk2(f0.x, f0.y), pk2(f0.z, f0.w), pk2(f1.x, f1.y), pk2(f1.z, f1.w) };
    U4 p1 = { pk2(f2.x, f2.y), pk2(f2.z, f2.w), pk2(f3.x, f3.y), pk2(f3.z, f3.w) };
    *(U4*)(&buf[sr * LDSP + sc])     = p0;
    *(U4*)(&buf[sr * LDSP + sc + 8]) = p1;
    __syncthreads();

    // prefetch next tile into registers while this tile computes
    if (d0 + 32 < ND) {
      const float* nsrc = src + d0 + 32;
      f0 = *(const F4*)(nsrc + 0);
      f1 = *(const F4*)(nsrc + 4);
      f2 = *(const F4*)(nsrc + 8);
      f3 = *(const F4*)(nsrc + 12);
    }

    Frag af;
    af.q[0] = *(const U4*)(&buf[arow * LDSP + ahalf]);
    af.q[1] = *(const U4*)(&buf[arow * LDSP + ahalf + 16]);

    Frag bg[4];
    #pragma unroll
    for (int nt = 0; nt < 4; ++nt) {
      const unsigned short* wp = wbf + (size_t)(nt * 16 + (lane & 15)) * ND + d0 + bhalf;
      bg[nt].q[0] = *(const U4*)(wp + 0);
      bg[nt].q[1] = *(const U4*)(wp + 16);
    }
    #pragma unroll
    for (int nt = 0; nt < 4; ++nt)
      acc[nt] = __builtin_amdgcn_wmma_f32_16x16x32_bf16(false, af.v, false, bg[nt].v,
                                                        (short)0, acc[nt], false, false);
  }

  const int len   = lens[b];
  const int mbase = l0 + wave * 16 + ahalf;   // C layout: VGPR r -> M = r (+8 for upper lanes)
  #pragma unroll
  for (int nt = 0; nt < 4; ++nt) {
    const int kcl = nt * 16 + (lane & 15);
    const float bias = ab[kcl];
    float vals[8];
    #pragma unroll
    for (int r = 0; r < 8; ++r) {
      const int gl = mbase + r;
      vals[r] = (gl < len) ? (acc[nt][r] + bias) : NEG_INF;
    }
    float* dst = logits + ((size_t)(b * NK + kcl) * NL + mbase);
    F4 o0 = { vals[0], vals[1], vals[2], vals[3] };
    F4 o1 = { vals[4], vals[5], vals[6], vals[7] };
    *(F4*)(dst + 0) = o0;
    *(F4*)(dst + 4) = o1;
  }
}

// ---------------- Kernel 2: per (b,k) softmax stats over L ----------------
__global__ __launch_bounds__(256) void k_stats(const float* __restrict__ logits,
                                               float* __restrict__ mx,
                                               float* __restrict__ inv) {
  const int bk  = blockIdx.x;
  const int tid = threadIdx.x;
  const float* p = logits + (size_t)bk * NL + tid * 16;
  F4 v[4];
  #pragma unroll
  for (int j = 0; j < 4; ++j) v[j] = *(const F4*)(p + j * 4);

  float m = NEG_INF;
  #pragma unroll
  for (int j = 0; j < 4; ++j)
    m = fmaxf(m, fmaxf(fmaxf(v[j].x, v[j].y), fmaxf(v[j].z, v[j].w)));

  __shared__ float red[256];
  red[tid] = m; __syncthreads();
  for (int s = 128; s > 0; s >>= 1) {
    if (tid < s) red[tid] = fmaxf(red[tid], red[tid + s]);
    __syncthreads();
  }
  m = red[0];
  __syncthreads();

  float s = 0.0f;
  #pragma unroll
  for (int j = 0; j < 4; ++j)
    s += __expf(v[j].x - m) + __expf(v[j].y - m) + __expf(v[j].z - m) + __expf(v[j].w - m);
  red[tid] = s; __syncthreads();
  for (int t = 128; t > 0; t >>= 1) {
    if (tid < t) red[tid] += red[tid + t];
    __syncthreads();
  }
  if (tid == 0) { mx[bk] = m; inv[bk] = 1.0f / red[0]; }
}

// ---------------- Kernel 3: vladT[b][d][k] = (aᵀ·feats)[k][d] - centroids[k][d] ----------------
__global__ __launch_bounds__(256) void k_vlad(const float* __restrict__ feats,
                                              const float* __restrict__ logits,
                                              const float* __restrict__ mx,
                                              const float* __restrict__ inv,
                                              const float* __restrict__ centT,
                                              float* __restrict__ vladT) {
  const int b   = blockIdx.x >> 3;
  const int d0  = (blockIdx.x & 7) * 128;
  const int tid  = threadIdx.x;
  const int wave = tid >> 5;
  const int lane = tid & 31;

  __shared__ unsigned short lsB[2][128 * LDSP];  // featsT: [d(128)][l(32)] bf16, double buffered
  __shared__ unsigned short lsA[2][64 * LDSP];   // a     : [k(64)][l(32)] bf16, double buffered

  // a-staging: thread -> (cluster, 8 l-values)
  const int ka = tid >> 2;
  const int lc = (tid & 3) * 8;
  const float amx  = mx[b * NK + ka];
  const float ainv = inv[b * NK + ka];
  const float* lrow = logits + (size_t)(b * NK + ka) * NL + lc;

  // feats-staging: thread -> (1 l-row, 16 d-cols), written transposed into LDS
  const int sl = tid >> 3;            // 0..31
  const int sc = (tid & 7) * 16;      // 0..112
  const float* fb = feats + ((size_t)b * NL + sl) * ND + d0 + sc;

  v8f acc[4];
  #pragma unroll
  for (int i = 0; i < 4; ++i) acc[i] = vzero();

  const int bd    = wave * 16 + (lane & 15);  // this wave's d column within tile
  const int bhalf = (lane >> 4) * 8;
  const int ahalf = (lane >> 4) * 8;

  // pipeline prologue: tile 0 in registers
  F4 a0 = *(const F4*)(lrow + 0);
  F4 a1 = *(const F4*)(lrow + 4);
  F4 g0 = *(const F4*)(fb + 0);
  F4 g1 = *(const F4*)(fb + 4);
  F4 g2 = *(const F4*)(fb + 8);
  F4 g3 = *(const F4*)(fb + 12);

  for (int l0 = 0; l0 < NL; l0 += 32) {
    const int bsel = (l0 >> 5) & 1;
    // stage softmax weights as bf16
    float w0 = __expf(a0.x - amx) * ainv, w1 = __expf(a0.y - amx) * ainv;
    float w2 = __expf(a0.z - amx) * ainv, w3 = __expf(a0.w - amx) * ainv;
    float w4 = __expf(a1.x - amx) * ainv, w5 = __expf(a1.y - amx) * ainv;
    float w6 = __expf(a1.z - amx) * ainv, w7 = __expf(a1.w - amx) * ainv;
    U4 pa = { pk2(w0, w1), pk2(w2, w3), pk2(w4, w5), pk2(w6, w7) };
    *(U4*)(&lsA[bsel][ka * LDSP + lc]) = pa;

    // stage feats tile transposed (row read from HBM, scattered bf16 into LDS)
    const float gv[16] = { g0.x, g0.y, g0.z, g0.w, g1.x, g1.y, g1.z, g1.w,
                           g2.x, g2.y, g2.z, g2.w, g3.x, g3.y, g3.z, g3.w };
    #pragma unroll
    for (int j = 0; j < 16; ++j)
      lsB[bsel][(sc + j) * LDSP + sl] = f2bf(gv[j]);
    __syncthreads();

    // prefetch next tile into registers while this tile computes
    if (l0 + 32 < NL) {
      a0 = *(const F4*)(lrow + l0 + 32);
      a1 = *(const F4*)(lrow + l0 + 36);
      const float* fr = fb + (size_t)(l0 + 32) * ND;
      g0 = *(const F4*)(fr + 0);
      g1 = *(const F4*)(fr + 4);
      g2 = *(const F4*)(fr + 8);
      g3 = *(const F4*)(fr + 12);
    }

    Frag bg;
    bg.q[0] = *(const U4*)(&lsB[bsel][bd * LDSP + bhalf]);
    bg.q[1] = *(const U4*)(&lsB[bsel][bd * LDSP + bhalf + 16]);
    Frag af[4];
    #pragma unroll
    for (int mt = 0; mt < 4; ++mt) {
      const int arow = mt * 16 + (lane & 15);
      af[mt].q[0] = *(const U4*)(&lsA[bsel][arow * LDSP + ahalf]);
      af[mt].q[1] = *(const U4*)(&lsA[bsel][arow * LDSP + ahalf + 16]);
    }
    #pragma unroll
    for (int mt = 0; mt < 4; ++mt)
      acc[mt] = __builtin_amdgcn_wmma_f32_16x16x32_bf16(false, af[mt].v, false, bg.v,
                                                        (short)0, acc[mt], false, false);
  }

  // epilogue: subtract centroids (s == 1 analytically), write (B, D, K)
  const int dcol = d0 + wave * 16 + (lane & 15);
  #pragma unroll
  for (int mt = 0; mt < 4; ++mt) {
    const int kbase = mt * 16 + ahalf;
    const float* cp = centT + (size_t)dcol * NK + kbase;
    F4 c0 = *(const F4*)(cp + 0);
    F4 c1 = *(const F4*)(cp + 4);
    F4 o0 = { acc[mt][0] - c0.x, acc[mt][1] - c0.y, acc[mt][2] - c0.z, acc[mt][3] - c0.w };
    F4 o1 = { acc[mt][4] - c1.x, acc[mt][5] - c1.y, acc[mt][6] - c1.z, acc[mt][7] - c1.w };
    float* dst = vladT + ((size_t)b * ND + dcol) * NK + kbase;
    *(F4*)(dst + 0) = o0;
    *(F4*)(dst + 4) = o1;
  }
}

// ---------------- Kernel 4: min over K, L2-normalize over D ----------------
__global__ __launch_bounds__(256) void k_minnorm(const float* __restrict__ vladT,
                                                 float* __restrict__ out) {
  const int b   = blockIdx.x;
  const int tid = threadIdx.x;
  float vals[4];
  #pragma unroll
  for (int j = 0; j < 4; ++j) {
    const float* p = vladT + ((size_t)b * ND + tid * 4 + j) * NK;
    float m = __builtin_inff();
    #pragma unroll
    for (int k = 0; k < NK; k += 4) {
      F4 v = *(const F4*)(p + k);
      m = fminf(m, fminf(fminf(v.x, v.y), fminf(v.z, v.w)));
    }
    vals[j] = m;
  }
  float ss = vals[0]*vals[0] + vals[1]*vals[1] + vals[2]*vals[2] + vals[3]*vals[3];
  __shared__ float red[256];
  red[tid] = ss; __syncthreads();
  for (int s = 128; s > 0; s >>= 1) {
    if (tid < s) red[tid] += red[tid + s];
    __syncthreads();
  }
  const float nrm  = fmaxf(sqrtf(red[0]), 1e-12f);
  const float rinv = 1.0f / nrm;
  F4 o = { vals[0]*rinv, vals[1]*rinv, vals[2]*rinv, vals[3]*rinv };
  *(F4*)(out + (size_t)b * ND + tid * 4) = o;
}

extern "C" void kernel_launch(void* const* d_in, const int* in_sizes, int n_in,
                              void* d_out, int out_size, void* d_ws, size_t ws_size,
                              hipStream_t stream) {
  (void)in_sizes; (void)n_in; (void)out_size; (void)ws_size;
  const float* feats = (const float*)d_in[0];
  const int*   lens  = (const int*)d_in[1];
  const float* aw    = (const float*)d_in[2];
  const float* ab    = (const float*)d_in[3];
  const float* cen   = (const float*)d_in[4];
  float* out = (float*)d_out;

  char* ws = (char*)d_ws;
  unsigned short* wbf   = (unsigned short*)(ws);                       // 128 KiB
  float*          centT = (float*)(ws + 131072);                       // 256 KiB
  float*          mx    = (float*)(ws + 131072 + 262144);              // 8 KiB
  float*          inv   = (float*)(ws + 131072 + 262144 + 8192);       // 8 KiB
  float*          logit = (float*)(ws + 131072 + 262144 + 16384);      // 32 MiB
  float*          vladT = (float*)(ws + 131072 + 262144 + 16384 +
                                   (size_t)NB * NK * NL * 4);          // 8 MiB

  k_prep   <<<(NK * ND) / 256,  256, 0, stream>>>(aw, cen, wbf, centT);
  k_logits <<<NB * (NL / 128),  256, 0, stream>>>(feats, lens, wbf, ab, logit);
  k_stats  <<<NB * NK,          256, 0, stream>>>(logit, mx, inv);
  k_vlad   <<<NB * (ND / 128),  256, 0, stream>>>(feats, logit, mx, inv, centT, vladT);
  k_minnorm<<<NB,               256, 0, stream>>>(vladT, out);
}
